// gatLayer_9122510537292
// MI455X (gfx1250) — matrix operand order repaired and checked by reading, not compiled, not run on hardware
//
#include <hip/hip_runtime.h>
#include <hip/hip_bf16.h>
#include <math.h>

// ---------------- types ----------------
typedef __attribute__((ext_vector_type(16))) __bf16 v16bf;
typedef __attribute__((ext_vector_type(8)))  float  v8f;
typedef __attribute__((ext_vector_type(4)))  unsigned int u32x4;
typedef __attribute__((ext_vector_type(8)))  int i32x8;
typedef __attribute__((ext_vector_type(4)))  int i32x4;

struct __attribute__((aligned(16))) U128 { unsigned int w[4]; };
union Frag { U128 q[2]; v16bf v; };

__device__ __forceinline__ unsigned short f2bf(float f) {
  union { float f; unsigned u; } x; x.f = f;
  unsigned r = x.u + 0x7FFFu + ((x.u >> 16) & 1u);
  return (unsigned short)(r >> 16);
}

// ---------------- TDM helper (Tensor Data Mover, gfx1250) ----------------
#if __has_builtin(__builtin_amdgcn_tensor_load_to_lds) && \
    __has_builtin(__builtin_amdgcn_s_wait_tensorcnt)
#define HAVE_TDM 1
#endif

#ifdef HAVE_TDM
// DMA a 128x32 bf16 tile (row stride lda elems) global -> LDS with a 16-byte
// pad after each 64-byte row => LDS pitch 80B, matching sA[128][40].
__device__ __forceinline__ void tdm_load_tileA(const unsigned short* gsrc,
                                               unsigned lds_byte_off, int lda) {
  unsigned long long ga = (unsigned long long)(size_t)gsrc;
  u32x4 g0;
  g0.x = 1u;                                        // count=1, user descriptor
  g0.y = lds_byte_off;                              // D#.lds_addr
  g0.z = (unsigned)(ga & 0xFFFFFFFFu);              // global_addr[31:0]
  g0.w = (unsigned)((ga >> 32) & 0x01FFFFFFu) | (2u << 30); // addr[56:32]|type=2
  const unsigned td0 = 32u, td1 = 128u;
  const unsigned long long s0 = (unsigned long long)(unsigned)lda; // dim0 stride
  i32x8 g1;
  g1[0] = (int)((1u << 16)      // data_size = 1 (2 bytes)
              | (1u << 20)      // pad_enable
              | (3u << 22)      // pad_interval: 16 DWORDs (64B) before pad
              | (3u << 25));    // pad_amount: 4 DWORDs (16B)
  g1[1] = (int)((td0 & 0xFFFFu) << 16);                 // tensor_dim0 lo16
  g1[2] = (int)(((td0 >> 16) & 0xFFFFu) | ((td1 & 0xFFFFu) << 16));
  g1[3] = (int)(((td1 >> 16) & 0xFFFFu) | (32u << 16)); // tile_dim0 = 32
  g1[4] = (int)128u;                                    // tile_dim1=128, tile_dim2=0
  g1[5] = (int)(unsigned)(s0 & 0xFFFFFFFFu);            // tensor_dim0_stride lo
  g1[6] = (int)(unsigned)((s0 >> 32) & 0xFFFFu);        // stride hi | dim1_stride lo=0
  g1[7] = 0;
  const i32x4 gz = {0, 0, 0, 0};
#if __has_include(<hip/amd_detail/amd_gfx1250_TDM.h>)
  const i32x8 gz8 = {0, 0, 0, 0, 0, 0, 0, 0};
  __builtin_amdgcn_tensor_load_to_lds(g0, g1, gz, gz, gz8, 0);  // clang-23 form
#else
  __builtin_amdgcn_tensor_load_to_lds(g0, g1, gz, gz, 0);       // ROCm 7.2 form
#endif
}
#endif

// ---------------- elementwise kernels ----------------
__global__ void k_cast_f2b(const float* __restrict__ src,
                           unsigned short* __restrict__ dst, long n) {
  long i = (long)blockIdx.x * blockDim.x + threadIdx.x;
  if (i < n) dst[i] = f2bf(src[i]);
}

// catb[b][n][0:256] = bf16(update[b][n][:]); catb[b][n][256:512] = curb[b][n][:]
__global__ void k_pack_cat(const float* __restrict__ upd,
                           const unsigned short* __restrict__ curb,
                           unsigned short* __restrict__ catb, long n) {
  long i = (long)blockIdx.x * blockDim.x + threadIdx.x;
  if (i >= n) return;
  long bn = i >> 9;          // b*N + n
  int  k  = (int)(i & 511);
  catb[i] = (k < 256) ? f2bf(upd[bn * 256 + k]) : curb[bn * 256 + (k - 256)];
}

// ---------------- generic WMMA GEMM ----------------
// C[z] (M x Nc) = sum_r A[z,r] (M x K) @ B[z,r] (K x Nc), bf16 in, f32 accum.
// Block tile 128x64, 8 waves (256 thr), wave tile 32x32 (2x2 WMMA), K-step 32.
struct GemmP {
  const unsigned short* A; const unsigned short* B;
  int aDiv, aMod; long aStride, aRStride;
  int bDiv, bMod; long bStride, bRStride;
  int M, Nc, K, lda, ldb, ldc;
  int rCount, mode;
  const float* bias; int biasDiv, biasMod; long biasStride;
  const float* rowmask; int rmDiv; long rmStride;
  unsigned short* Cb; float* Cf; long cStride;
  const float* Cadd; long caStride;                      // mode 4
  const float* gUpd; const float* gCur; float* gOut; long gStride; // mode 5
};

__launch_bounds__(256)
__global__ void k_gemm(GemmP p) {
  __shared__ unsigned short sA[128][40];  // row-major A tile, 80B pitch
  __shared__ unsigned short sB[64][40];   // B tile stored transposed: sB[col][k]

  const int tid  = threadIdx.x;
  const int lane = tid & 31;
  const int w    = tid >> 5;            // 0..7
  const int wm   = (w & 3) * 32;        // wave M offset in block tile
  const int wn   = (w >> 2) * 32;       // wave N offset
  const int half = lane >> 4;
  const int l16  = lane & 15;
  const int z    = blockIdx.z;
  const int bm   = blockIdx.y * 128;
  const int bn   = blockIdx.x * 64;

  const unsigned short* Abase = p.A + (long)((z / p.aDiv) % p.aMod) * p.aStride;
  const unsigned short* Bbase = p.B + (long)((z / p.bDiv) % p.bMod) * p.bStride;

  v8f acc[2][2];
  const v8f vzero = {0.f,0.f,0.f,0.f,0.f,0.f,0.f,0.f};
  acc[0][0] = vzero; acc[0][1] = vzero; acc[1][0] = vzero; acc[1][1] = vzero;

  for (int r = 0; r < p.rCount; ++r) {
    const unsigned short* Ab = Abase + (long)r * p.aRStride;
    const unsigned short* Bb = Bbase + (long)r * p.bRStride;
    for (int k0 = 0; k0 < p.K; k0 += 32) {
      __syncthreads();
#ifdef HAVE_TDM
      if (w == 0) {  // wave 0 issues one TDM DMA for the whole A tile
        tdm_load_tileA(Ab + (long)bm * p.lda + k0,
                       (unsigned)(size_t)(void*)&sA[0][0], p.lda);
      }
#else
      { // stage A: 128x32 (fallback path)
        int row = tid >> 1, c0 = (tid & 1) * 16;
        const unsigned short* g = Ab + (long)(bm + row) * p.lda + k0 + c0;
        *(U128*)&sA[row][c0]     = *(const U128*)g;
        *(U128*)&sA[row][c0 + 8] = *(const U128*)(g + 8);
        if (k0 + 32 < p.K) __builtin_prefetch(g + 32, 0, 3);
      }
#endif
      { // stage B: 32x64, transposed into LDS (all 256 threads)
        int kr = tid >> 3, c0 = (tid & 7) * 8;
        const unsigned short* g = Bb + (long)(k0 + kr) * p.ldb + bn + c0;
        U128 v = *(const U128*)g;
        const unsigned short* e = (const unsigned short*)&v;
#pragma unroll
        for (int i = 0; i < 8; ++i) sB[c0 + i][kr] = e[i];
        if (k0 + 32 < p.K) __builtin_prefetch(g + 32 * p.ldb, 0, 3);
      }
#ifdef HAVE_TDM
      if (w == 0) __builtin_amdgcn_s_wait_tensorcnt(0);
#endif
      __syncthreads();

      Frag a[2], b[2];
#pragma unroll
      for (int i = 0; i < 2; ++i) { // A frag: lanes 0-15 rows, K interleave per ISA
        int row = wm + i * 16 + l16;
        a[i].q[0] = *(const U128*)&sA[row][half * 8];
        a[i].q[1] = *(const U128*)&sA[row][16 + half * 8];
      }
#pragma unroll
      for (int j = 0; j < 2; ++j) { // B frag: lane = col, K interleave per ISA (same as A)
        int col = wn + j * 16 + l16;
        b[j].q[0] = *(const U128*)&sB[col][half * 8];
        b[j].q[1] = *(const U128*)&sB[col][16 + half * 8];
      }
#pragma unroll
      for (int i = 0; i < 2; ++i)
#pragma unroll
        for (int j = 0; j < 2; ++j)
          acc[i][j] = __builtin_amdgcn_wmma_f32_16x16x32_bf16(
              false, a[i].v, false, b[j].v, (short)0, acc[i][j], false, false);
    }
  }

  const float* biasP = p.bias ? p.bias + (long)((z / p.biasDiv) % p.biasMod) * p.biasStride : nullptr;
  const float* rmP   = p.rowmask ? p.rowmask + (long)(z / p.rmDiv) * p.rmStride : nullptr;
  const long coff = (long)z * p.cStride;
#pragma unroll
  for (int i = 0; i < 2; ++i)
#pragma unroll
    for (int j = 0; j < 2; ++j)
#pragma unroll
      for (int e = 0; e < 8; ++e) {
        int row = bm + wm + i * 16 + e + half * 8;   // C layout: VGPR e, halves M / M+8
        int col = bn + wn + j * 16 + l16;
        float v = acc[i][j][e];
        long idx = coff + (long)row * p.ldc + col;
        switch (p.mode) {
          case 0: p.Cb[idx] = f2bf(v); break;
          case 1: p.Cb[idx] = f2bf((v + biasP[col]) * rmP[row]); break;
          case 2: p.Cf[idx] = (v + biasP[col]) * rmP[row]; break;
          case 4: p.Cf[idx] = v + p.Cadd[(long)z * p.caStride + (long)row * p.ldc + col]; break;
          case 5: {
            float g = 1.0f / (1.0f + __expf(-(v + biasP[col])));
            g *= rmP[row];
            long gi = (long)z * p.gStride + (long)row * p.ldc + col;
            float u = p.gUpd[gi], c = p.gCur[gi];
            p.gOut[gi] = g * tanhf(u) + (1.0f - g) * c;
          } break;
        }
      }
}

// ---------------- fused attention: S = q@cur^T, adj-mask, row softmax ----------------
// Block = 256 thr (8 waves) computes 16 rows x 1024 cols for one (b,r).
__launch_bounds__(256)
__global__ void k_att(const unsigned short* __restrict__ q,
                      const unsigned short* __restrict__ curb,
                      const float* __restrict__ adj,
                      unsigned short* __restrict__ att, float scale) {
  const int N = 1024, D = 256;
  const int tid  = threadIdx.x;
  const int lane = tid & 31, w = tid >> 5;
  const int half = lane >> 4, l16 = lane & 15;
  const int rb = blockIdx.x * 16;          // row block
  const int zr = blockIdx.y;               // b*R + r
  const int b  = zr >> 1;

  const unsigned short* Q = q    + (long)zr * N * D;
  const unsigned short* C = curb + (long)b  * N * D;
  const float* A          = adj  + (long)zr * N * N;
  unsigned short* O       = att  + (long)zr * N * N;

  v8f acc[8];
  const v8f vzero = {0.f,0.f,0.f,0.f,0.f,0.f,0.f,0.f};
#pragma unroll
  for (int t = 0; t < 8; ++t) acc[t] = vzero;

#pragma unroll
  for (int k0 = 0; k0 < D; k0 += 32) {
    Frag a;
    {
      const unsigned short* g = Q + (long)(rb + l16) * D + k0;
      a.q[0] = *(const U128*)(g + half * 8);
      a.q[1] = *(const U128*)(g + 16 + half * 8);
    }
#pragma unroll
    for (int t = 0; t < 8; ++t) {
      Frag bf;
      int col = w * 128 + t * 16 + l16;    // column of cur^T == row of cur
      const unsigned short* g = C + (long)col * D + k0;
      bf.q[0] = *(const U128*)(g + half * 8);
      bf.q[1] = *(const U128*)(g + 16 + half * 8);
      acc[t] = __builtin_amdgcn_wmma_f32_16x16x32_bf16(
          false, a.v, false, bf.v, (short)0, acc[t], false, false);
    }
  }

  // mask with adj, keep per-lane values
  float val[8][8];
#pragma unroll
  for (int t = 0; t < 8; ++t)
#pragma unroll
    for (int e = 0; e < 8; ++e) {
      int row = rb + e + half * 8;
      int col = w * 128 + t * 16 + l16;
      float av = A[(long)row * N + col];
      val[t][e] = (av > 0.f) ? acc[t][e] * scale : -9e15f;
    }

  __shared__ float red[16][8];

  // row max: per-lane over t, shfl within 16-lane group, then cross-wave via LDS
  float rmax[8];
#pragma unroll
  for (int e = 0; e < 8; ++e) {
    float m = val[0][e];
#pragma unroll
    for (int t = 1; t < 8; ++t) m = fmaxf(m, val[t][e]);
#pragma unroll
    for (int off = 8; off >= 1; off >>= 1) m = fmaxf(m, __shfl_xor(m, off, 32));
    rmax[e] = m;
  }
  if (l16 == 0) {
#pragma unroll
    for (int e = 0; e < 8; ++e) red[e + half * 8][w] = rmax[e];
  }
  __syncthreads();
  float rowmax[8];
#pragma unroll
  for (int e = 0; e < 8; ++e) {
    int rr = e + half * 8;
    float m = red[rr][0];
#pragma unroll
    for (int ww = 1; ww < 8; ++ww) m = fmaxf(m, red[rr][ww]);
    rowmax[e] = m;
  }
  __syncthreads();

  // exp + row sum
  float rsum[8];
#pragma unroll
  for (int e = 0; e < 8; ++e) {
    float s = 0.f;
#pragma unroll
    for (int t = 0; t < 8; ++t) {
      val[t][e] = __expf(val[t][e] - rowmax[e]);
      s += val[t][e];
    }
#pragma unroll
    for (int off = 8; off >= 1; off >>= 1) s += __shfl_xor(s, off, 32);
    rsum[e] = s;
  }
  if (l16 == 0) {
#pragma unroll
    for (int e = 0; e < 8; ++e) red[e + half * 8][w] = rsum[e];
  }
  __syncthreads();
#pragma unroll
  for (int e = 0; e < 8; ++e) {
    int rr = e + half * 8;
    float s = 0.f;
#pragma unroll
    for (int ww = 0; ww < 8; ++ww) s += red[rr][ww];
    float inv = 1.0f / s;
    int row = rb + rr;
#pragma unroll
    for (int t = 0; t < 8; ++t) {
      int col = w * 128 + t * 16 + l16;
      O[(long)row * N + col] = f2bf(val[t][e] * inv);
    }
  }
}

// ---------------- host ----------------
static inline void launch_cast(const float* s, unsigned short* d, long n, hipStream_t st) {
  k_cast_f2b<<<dim3((unsigned)((n + 255) / 256)), dim3(256), 0, st>>>(s, d, n);
}

extern "C" void kernel_launch(void* const* d_in, const int* in_sizes, int n_in,
                              void* d_out, int out_size, void* d_ws, size_t ws_size,
                              hipStream_t stream) {
  const int B = 8, N = 1024, D = 256, R = 2, P = 256, HOPS = 3;
  const float* x    = (const float*)d_in[0];
  const float* mask = (const float*)d_in[1];
  const float* adj  = (const float*)d_in[2];
  const float* Wa   = (const float*)d_in[3];
  const float* Wr   = (const float*)d_in[4];
  const float* br   = (const float*)d_in[5];
  const float* Ws   = (const float*)d_in[6];
  const float* bs   = (const float*)d_in[7];
  const float* Wg   = (const float*)d_in[8];
  const float* bg   = (const float*)d_in[9];

  char* wsp = (char*)d_ws;
  auto alloc = [&](size_t bytes) { char* p = wsp; wsp += (bytes + 255) & ~(size_t)255; return p; };

  float*          cur    = (float*)alloc((size_t)B * N * D * 4);
  unsigned short* curb   = (unsigned short*)alloc((size_t)B * N * D * 2);
  unsigned short* qb     = (unsigned short*)alloc((size_t)B * R * N * D * 2);
  unsigned short* nb     = (unsigned short*)alloc((size_t)B * R * N * P * 2);
  float*          su     = (float*)alloc((size_t)B * N * P * 4);
  unsigned short* attb   = (unsigned short*)alloc((size_t)B * R * N * N * 2);
  unsigned short* adjb   = (unsigned short*)alloc((size_t)B * R * N * N * 2);
  unsigned short* aggb   = (unsigned short*)alloc((size_t)B * R * N * P * 2);
  float*          upd    = (float*)alloc((size_t)B * N * P * 4);
  unsigned short* catb   = (unsigned short*)alloc((size_t)B * N * (P + D) * 2);
  unsigned short* Wab    = (unsigned short*)alloc((size_t)R * D * D * 2);
  unsigned short* Wrb    = (unsigned short*)alloc((size_t)R * D * P * 2);
  unsigned short* Wsb    = (unsigned short*)alloc((size_t)D * P * 2);
  unsigned short* Wgb    = (unsigned short*)alloc((size_t)(P + D) * P * 2);

  // one-time conversions
  hipMemcpyAsync(cur, x, (size_t)B * N * D * 4, hipMemcpyDeviceToDevice, stream);
  launch_cast(adj, adjb, (long)B * R * N * N, stream);
  launch_cast(Wa, Wab, (long)R * D * D, stream);
  launch_cast(Wr, Wrb, (long)R * D * P, stream);
  launch_cast(Ws, Wsb, (long)D * P, stream);
  launch_cast(Wg, Wgb, (long)(P + D) * P, stream);

  const int BIG = 1 << 30;
  const dim3 blk(256);
  const dim3 grd16(256 / 64, 1024 / 128, B * R);
  const dim3 grd8(256 / 64, 1024 / 128, B);

  for (int hop = 0; hop < HOPS; ++hop) {
    launch_cast(cur, curb, (long)B * N * D, stream);

    GemmP p{};
    p.M = N; p.Nc = D; p.rCount = 1;

    // q[b,r] = curb[b] @ Wab[r]
    p.A = curb; p.aDiv = R; p.aMod = BIG; p.aStride = (long)N * D; p.aRStride = 0;
    p.B = Wab;  p.bDiv = 1; p.bMod = R;   p.bStride = (long)D * D; p.bRStride = 0;
    p.K = D; p.lda = D; p.ldb = D; p.ldc = D;
    p.mode = 0; p.Cb = qb; p.cStride = (long)N * D;
    k_gemm<<<grd16, blk, 0, stream>>>(p);

    // nb[b,r] = (curb[b] @ Wrb[r] + br[r]) * mask[b]
    p.B = Wrb; p.bStride = (long)D * P;
    p.mode = 1; p.Cb = nb; p.cStride = (long)N * P;
    p.bias = br; p.biasDiv = 1; p.biasMod = R; p.biasStride = P;
    p.rowmask = mask; p.rmDiv = R; p.rmStride = N;
    k_gemm<<<grd16, blk, 0, stream>>>(p);

    // su[b] = (curb[b] @ Wsb + bs) * mask[b]
    p.aDiv = 1; p.B = Wsb; p.bMod = 1; p.bStride = 0;
    p.mode = 2; p.Cf = su; p.Cb = nullptr; p.cStride = (long)N * P;
    p.bias = bs; p.biasMod = 1; p.biasStride = 0;
    p.rmDiv = 1;
    k_gemm<<<grd8, blk, 0, stream>>>(p);

    // att = softmax(mask_adj(q @ cur^T * scale))
    k_att<<<dim3(N / 16, B * R), blk, 0, stream>>>(qb, curb, adj, attb, 0.0625f);

    // agg[b,r] = adjb[b,r] @ nb[b,r]
    GemmP g{};
    g.M = N; g.Nc = P; g.rCount = 1;
    g.A = adjb; g.aDiv = 1; g.aMod = BIG; g.aStride = (long)N * N;
    g.B = nb;   g.bDiv = 1; g.bMod = BIG; g.bStride = (long)N * P;
    g.K = N; g.lda = N; g.ldb = P; g.ldc = P;
    g.mode = 0; g.Cb = aggb; g.cStride = (long)N * P;
    k_gemm<<<grd16, blk, 0, stream>>>(g);

    // update[b] = sum_r att[b,r] @ agg[b,r] + su[b]
    GemmP u{};
    u.M = N; u.Nc = P; u.rCount = R;
    u.A = attb; u.aDiv = 1; u.aMod = BIG; u.aStride = (long)R * N * N; u.aRStride = (long)N * N;
    u.B = aggb; u.bDiv = 1; u.bMod = BIG; u.bStride = (long)R * N * P; u.bRStride = (long)N * P;
    u.K = N; u.lda = N; u.ldb = P; u.ldc = P;
    u.mode = 4; u.Cf = upd; u.cStride = (long)N * P;
    u.Cadd = su; u.caStride = (long)N * P;
    k_gemm<<<grd8, blk, 0, stream>>>(u);

    // catb = [bf16(update), curb]
    k_pack_cat<<<dim3((unsigned)(((long)B * N * (P + D) + 255) / 256)), blk, 0, stream>>>(
        upd, curb, catb, (long)B * N * (P + D));

    // gate + blend, writes cur in place
    GemmP q5{};
    q5.M = N; q5.Nc = P; q5.rCount = 1;
    q5.A = catb; q5.aDiv = 1; q5.aMod = BIG; q5.aStride = (long)N * (P + D);
    q5.B = Wgb;  q5.bDiv = 1; q5.bMod = 1;   q5.bStride = 0;
    q5.K = P + D; q5.lda = P + D; q5.ldb = P; q5.ldc = P;
    q5.mode = 5;
    q5.bias = bg; q5.biasDiv = 1; q5.biasMod = 1; q5.biasStride = 0;
    q5.rowmask = mask; q5.rmDiv = 1; q5.rmStride = N;
    q5.gUpd = upd; q5.gCur = cur; q5.gOut = cur; q5.gStride = (long)N * P;
    k_gemm<<<grd8, blk, 0, stream>>>(q5);
  }

  hipMemcpyAsync(d_out, cur, (size_t)out_size * sizeof(float),
                 hipMemcpyDeviceToDevice, stream);
}
